// SHDSNNv8_15831249453640
// MI455X (gfx1250) — matrix operand order repaired and checked by reading, not compiled, not run on hardware
//
#include <hip/hip_runtime.h>

// ---------------------------------------------------------------------------
// MI455X (gfx1250) adLIF SNN forward.
// bf16 WMMA GEMMs (v_wmma_f32_16x16x32_bf16), table-order k in bf16 A/B fragments,
// LDS ping-pong + global->reg->LDS software pipeline, fused fp32 neuron
// updates in the epilogue.
// ---------------------------------------------------------------------------

typedef __attribute__((ext_vector_type(16))) __bf16 v16bf;
typedef __attribute__((ext_vector_type(8)))  __bf16 v8bf;
typedef __attribute__((ext_vector_type(8)))  float  v8f;

#define BB   256    // batch
#define TT   250    // timesteps
#define NIN  700
#define H1   1024
#define H2   512
#define NOUT 20
#define K1   1728   // 704 (x bf16, padded) + 1024 (recurrent s1)
#define K2   1536   // 1024 (s1)            +  512 (recurrent s2)
#define KS1  704
#define KS2  1024

#define BM 64
#define BN 128
#define BK 32
#define LDA 40      // LDS row strides (elements); multiple of 8 -> 16B aligned frags
#define LDB 40

union V16 { v16bf v; v8bf h[2]; };

// ---------------------------------------------------------------------------
// Fused GEMM + adLIF update.
//   I[m,n] = sum_k Asrc[m,k] * Bp[n,k]     (both bf16, K-contiguous rows)
//   Asrc = [ drive | prev spikes ] split at k_split; prev spikes also feed
//   the soft reset. New spikes are scattered into consumer A-buffers.
// ---------------------------------------------------------------------------
__global__ __launch_bounds__(256) void snn_layer_gemm(
    const __bf16* __restrict__ Asrc, int K_tot, int k_split, int N_tot,
    const __bf16* __restrict__ Bp,            // packed weights [N_tot][K_tot]
    const float* __restrict__ alpha, const float* __restrict__ rho,
    const float* __restrict__ beta_a,
    float* __restrict__ v_st, float* __restrict__ a_st,
    __bf16* __restrict__ sd0, int sd0_stride, int sd0_off,   // spike dest 0
    __bf16* __restrict__ sd1, int sd1_stride, int sd1_off)   // spike dest 1 (opt)
{
    __shared__ __bf16 As[2][BM * LDA];
    __shared__ __bf16 Bs[2][BN * LDB];

    const int tid = threadIdx.x;
    const int m0  = blockIdx.x * BM;
    const int n0  = blockIdx.y * BN;

    const int wave = tid >> 5;          // wave32
    const int lane = tid & 31;
    const int hi   = lane >> 4;
    const int l15  = lane & 15;
    const int wm   = (wave >> 2) * 32;  // 2x4 wave grid, 32x32 tile per wave
    const int wn   = (wave & 3) * 32;

    v8f acc[2][2] = {};

    const int ar = tid >> 2;            // A staging: 64 rows x 4 chunks of 8
    const int ac = (tid & 3) * 8;
    const int br = tid >> 1;            // B staging: 128 rows x 2 chunks of 16
    const int bc = (tid & 1) * 16;

    const __bf16* gA = Asrc + (size_t)(m0 + ar) * K_tot + ac;
    const __bf16* gB = Bp   + (size_t)(n0 + br) * K_tot + bc;

    // global -> register prefetch (issued before WMMAs of the current tile)
    v8bf rA, rB0, rB1;
    auto ld = [&](int kt) {
        rA  = *(const v8bf*)(gA + kt);
        rB0 = *(const v8bf*)(gB + kt);
        rB1 = *(const v8bf*)(gB + kt + 8);
        if (kt + BK < K_tot)
            __builtin_prefetch(gB + kt + BK, 0, 3);   // global_prefetch_b8
    };
    // register -> LDS store (after WMMAs; wait lands off the critical path)
    auto st = [&](int buf) {
        *(v8bf*)&As[buf][ar * LDA + ac]     = rA;
        *(v8bf*)&Bs[buf][br * LDB + bc]     = rB0;
        *(v8bf*)&Bs[buf][br * LDB + bc + 8] = rB1;
    };

    ld(0);
    st(0);
    const int NK = K_tot / BK;
    for (int it = 0; it < NK; ++it) {
        __syncthreads();                 // tile `it` visible; buf it^1 free
        const int cur = it & 1;
        const bool more = (it + 1 < NK);
        if (more) ld((it + 1) * BK);     // next-tile loads in flight over WMMAs

        // fragments per ISA 16-bit layouts
        V16 af[2], bw[2];
        #pragma unroll
        for (int ti = 0; ti < 2; ++ti) {
            // A 16x32: lane<16 -> K{0..7,16..23}; lane>=16 -> K{8..15,24..31}
            const int row = wm + ti * 16 + l15;
            const int k0  = hi * 8;
            af[ti].h[0] = *(const v8bf*)&As[cur][row * LDA + k0];
            af[ti].h[1] = *(const v8bf*)&As[cur][row * LDA + k0 + 16];
        }
        #pragma unroll
        for (int tj = 0; tj < 2; ++tj) {
            // B 32x16: lane<16 -> K{0..7,16..23}; lane>=16 -> K{8..15,24..31}
            const int nr = wn + tj * 16 + l15;
            const int k0 = hi * 8;
            bw[tj].h[0] = *(const v8bf*)&Bs[cur][nr * LDB + k0];
            bw[tj].h[1] = *(const v8bf*)&Bs[cur][nr * LDB + k0 + 16];
        }
        #pragma unroll
        for (int ti = 0; ti < 2; ++ti)
            #pragma unroll
            for (int tj = 0; tj < 2; ++tj)
                acc[ti][tj] = __builtin_amdgcn_wmma_f32_16x16x32_bf16(
                    false, af[ti].v, false, bw[tj].v,
                    (short)0, acc[ti][tj], false, false);

        if (more) st(cur ^ 1);           // commit next tile to the other buffer
    }

    // ---- fused adLIF epilogue; C/D layout: VGPR r -> M=r (+8 on lane hi) ----
    #pragma unroll
    for (int ti = 0; ti < 2; ++ti) {
        #pragma unroll
        for (int tj = 0; tj < 2; ++tj) {
            #pragma unroll
            for (int r = 0; r < 8; ++r) {
                const int m = m0 + wm + ti * 16 + hi * 8 + r;
                const int n = n0 + wn + tj * 16 + l15;
                const float I  = acc[ti][tj][r];
                const float sp = (float)Asrc[(size_t)m * K_tot + k_split + n];
                const float al = alpha[n];
                const size_t sx = (size_t)m * N_tot + n;
                float v = v_st[sx];
                float a = a_st[sx];
                v = al * (v - sp) + (1.f - al) * (I - a);    // thr = 1.0
                const float sn = (v >= 1.f) ? 1.f : 0.f;     // Heaviside forward
                a = rho[n] * a + beta_a[n] * sn;
                v_st[sx] = v;
                a_st[sx] = a;
                const __bf16 sb = (__bf16)sn;                // 0/1 exact in bf16
                sd0[(size_t)m * sd0_stride + sd0_off + n] = sb;
                if (sd1) sd1[(size_t)m * sd1_stride + sd1_off + n] = sb;
            }
        }
    }
}

// Per-step x_t fp32 -> bf16 into the L1 A-buffer (cols 0..703, zero padded).
__global__ __launch_bounds__(256) void snn_xcvt(
    const float* __restrict__ xt, long row_stride, __bf16* __restrict__ dst)
{
    const int gid = blockIdx.x * 256 + threadIdx.x;
    if (gid >= BB * KS1) return;
    const int m = gid / KS1, k = gid % KS1;
    const float f = (k < NIN) ? xt[(size_t)m * row_stride + k] : 0.f;
    dst[(size_t)m * K1 + k] = (__bf16)f;
}

// ---------------------------------------------------------------------------
// Leaky-integrator readout: I_out = s2 @ Wout^T; v_out recurrence + time sum.
// ---------------------------------------------------------------------------
__global__ __launch_bounds__(256) void snn_readout(
    const __bf16* __restrict__ s2, int s2_stride,
    const __bf16* __restrict__ WoB, const float* __restrict__ beta_out,
    float* __restrict__ vout, float* __restrict__ accum)
{
    const int gid = blockIdx.x * 256 + threadIdx.x;
    if (gid >= BB * NOUT) return;
    const int b = gid / NOUT;
    const int o = gid % NOUT;
    const __bf16* sr = s2  + (size_t)b * s2_stride;
    const __bf16* wr = WoB + (size_t)o * H2;
    float acc = 0.f;
    for (int k = 0; k < H2; k += 8) {
        const v8bf a = *(const v8bf*)(sr + k);
        const v8bf w = *(const v8bf*)(wr + k);
        #pragma unroll
        for (int j = 0; j < 8; ++j) acc += (float)a[j] * (float)w[j];
    }
    const float bo = beta_out[o];
    float vo = vout[gid];
    vo = bo * vo + (1.f - bo) * acc;
    vout[gid] = vo;
    accum[gid] += vo;
}

__global__ void snn_final(const float* __restrict__ accum, float* __restrict__ out) {
    const int gid = blockIdx.x * 256 + threadIdx.x;
    if (gid < BB * NOUT) out[gid] = accum[gid] * (1.f / (float)TT);
}

// Pack fp32 weights -> bf16 [N][K] with recurrent block concatenated on K.
__global__ void snn_pack(__bf16* __restrict__ dst,
                         const float* __restrict__ W, const float* __restrict__ Wrec,
                         int c1_valid, int k_split, int Ktot, int wrec_cols, int total)
{
    const int i = blockIdx.x * 256 + threadIdx.x;
    if (i >= total) return;
    const int n = i / Ktot, k = i % Ktot;
    float v;
    if (k < k_split) v = (k < c1_valid) ? W[(size_t)n * c1_valid + k] : 0.f;
    else             v = Wrec[(size_t)n * wrec_cols + (k - k_split)];
    dst[i] = (__bf16)v;
}

// ---------------------------------------------------------------------------
extern "C" void kernel_launch(void* const* d_in, const int* in_sizes, int n_in,
                              void* d_out, int out_size, void* d_ws, size_t ws_size,
                              hipStream_t stream)
{
    (void)in_sizes; (void)n_in; (void)out_size; (void)ws_size;
    const float* x      = (const float*)d_in[0];
    const float* W1     = (const float*)d_in[1];
    const float* Wrec1  = (const float*)d_in[2];
    const float* W2     = (const float*)d_in[3];
    const float* Wrec2  = (const float*)d_in[4];
    const float* Wout   = (const float*)d_in[5];
    const float* alpha1 = (const float*)d_in[6];
    const float* rho1   = (const float*)d_in[7];
    const float* beta1  = (const float*)d_in[8];
    const float* alpha2 = (const float*)d_in[9];
    const float* rho2   = (const float*)d_in[10];
    const float* beta2  = (const float*)d_in[11];
    const float* betao  = (const float*)d_in[12];

    char* ws = (char*)d_ws;
    size_t off = 0;
    auto alloc = [&](size_t bytes) {
        char* p = ws + off; off += (bytes + 255) & ~(size_t)255; return p;
    };
    __bf16* WB1 = (__bf16*)alloc((size_t)H1 * K1 * 2);      // ~3.5 MB, L2-resident
    __bf16* WB2 = (__bf16*)alloc((size_t)H2 * K2 * 2);      // ~1.6 MB
    __bf16* WoB = (__bf16*)alloc((size_t)NOUT * H2 * 2);
    char* state = ws + off;
    float* v1    = (float*)alloc((size_t)BB * H1 * 4);
    float* a1    = (float*)alloc((size_t)BB * H1 * 4);
    float* v2    = (float*)alloc((size_t)BB * H2 * 4);
    float* a2    = (float*)alloc((size_t)BB * H2 * 4);
    float* vout  = (float*)alloc((size_t)BB * NOUT * 4);
    float* accum = (float*)alloc((size_t)BB * NOUT * 4);
    __bf16* A1buf[2] = { (__bf16*)alloc((size_t)BB * K1 * 2),   // [x | s1_prev]
                         (__bf16*)alloc((size_t)BB * K1 * 2) };
    __bf16* A2buf[2] = { (__bf16*)alloc((size_t)BB * K2 * 2),   // [s1 | s2_prev]
                         (__bf16*)alloc((size_t)BB * K2 * 2) };
    const size_t state_bytes = (size_t)(ws + off - state);

    // Zero recurrent state + spike regions (graph-capturable).
    hipMemsetAsync(state, 0, state_bytes, stream);

    { const int tot = H1 * K1;
      snn_pack<<<(tot + 255) / 256, 256, 0, stream>>>(WB1, W1, Wrec1, NIN, KS1, K1, H1, tot); }
    { const int tot = H2 * K2;
      snn_pack<<<(tot + 255) / 256, 256, 0, stream>>>(WB2, W2, Wrec2, H1, KS2, K2, H2, tot); }
    { const int tot = NOUT * H2;
      snn_pack<<<(tot + 255) / 256, 256, 0, stream>>>(WoB, Wout, Wout, H2, H2, H2, H2, tot); }

    for (int t = 0; t < TT; ++t) {
        const int c = t & 1, nx = (t + 1) & 1;   // step-parity A-buffers
        // x_t -> bf16 drive region of this step's L1 A-buffer
        snn_xcvt<<<(BB * KS1 + 255) / 256, 256, 0, stream>>>(
            x + (size_t)t * NIN, (long)TT * NIN, A1buf[c]);
        // Layer 1: I1 = [x_t | s1(t-1)] @ WB1^T ; s1(t) -> L2 A-buf + next L1 A-buf
        snn_layer_gemm<<<dim3(BB / BM, H1 / BN), 256, 0, stream>>>(
            A1buf[c], K1, KS1, H1, WB1,
            alpha1, rho1, beta1, v1, a1,
            A2buf[c],  K2, 0,        // s1(t) as L2 drive
            A1buf[nx], K1, KS1);     // s1(t) as next-step recurrent operand
        // Layer 2: I2 = [s1(t) | s2(t-1)] @ WB2^T ; s2(t) -> next L2 A-buf
        snn_layer_gemm<<<dim3(BB / BM, H2 / BN), 256, 0, stream>>>(
            A2buf[c], K2, KS2, H2, WB2,
            alpha2, rho2, beta2, v2, a2,
            A2buf[nx], K2, KS2,
            nullptr,   0,  0);
        // Readout + leaky integration + time accumulation (reads s2(t))
        snn_readout<<<(BB * NOUT + 255) / 256, 256, 0, stream>>>(
            A2buf[nx] + KS2, K2, WoB, betao, vout, accum);
    }
    snn_final<<<(BB * NOUT + 255) / 256, 256, 0, stream>>>(accum, (float*)d_out);
}
